// GraphLayer_31533649887818
// MI455X (gfx1250) — matrix operand order repaired and checked by reading, not compiled, not run on hardware
//
#include <hip/hip_runtime.h>
#include <hip/hip_bf16.h>

// ---------------------------------------------------------------------------
// Problem constants (match the JAX reference)
// ---------------------------------------------------------------------------
#define KDIM 128
#define EDIM 16
#define NPQ_TOT (64 * 1024)
#define NPV_TOT (64 * 256)
#define NSL_TOT (64 * 1)
#define LDA 136   // padded bf16 LDS row stride (272B = 68 dwords -> conflict-free)

typedef __attribute__((ext_vector_type(16))) __bf16 v16bf;
typedef __attribute__((ext_vector_type(8)))  __bf16 v8bf;
typedef __attribute__((ext_vector_type(8)))  float  v8f;

union V16U { v16bf v; struct { v8bf lo; v8bf hi; } h; };

// ---------------------------------------------------------------------------
// WMMA GEMM:  Y[nrows,128] (+)= X[nrows,128] @ W[128,128] (+ bias)
// block = 256 threads (8 waves). W is transposed to bf16 LDS once per block,
// then the block sweeps 4 chunks of 32 rows (128 rows/block) against it.
// Per chunk each wave computes two 16x16 C tiles = 8 v_wmma_f32_16x16x32_bf16.
// Compile-time epilogue variants keep the store path branch-free.
// ---------------------------------------------------------------------------
template <int BETA, int HASBIAS>
__global__ __launch_bounds__(256)
void gemm128_wmma(const float* __restrict__ X, const float* __restrict__ W,
                  const float* __restrict__ bias, float* __restrict__ Y,
                  int nrows)
{
    __shared__ __bf16 sA[32 * LDA];    // X chunk, row-major [r][k]
    __shared__ __bf16 sW[128 * LDA];   // W transposed     [n][k]

    const int  tid     = threadIdx.x;
    const long rowbase = (long)blockIdx.x * 128;

    // Stage W (transposed) into LDS as bf16 -- once per block.
    for (int i = tid; i < KDIM * KDIM; i += 256) {
        int k = i >> 7, n = i & 127;
        sW[n * LDA + k] = (__bf16)W[i];
    }

    const int wave = tid >> 5;
    const int lane = tid & 31;
    const int rt   = wave >> 2;        // row tile within chunk: 0..1
    const int cp   = wave & 3;         // column pair: 0..3
    const int n0a  = cp * 32;
    const int n0b  = cp * 32 + 16;
    const int half = lane >> 4;        // lane group (ISA 16-bit A/B layout)
    const int l16  = lane & 15;

    const float biasA = HASBIAS ? bias[n0a + l16] : 0.0f;
    const float biasB = HASBIAS ? bias[n0b + l16] : 0.0f;

    for (int chunk = 0; chunk < 4; ++chunk) {
        const long row0blk = rowbase + chunk * 32;
        if (row0blk >= nrows) break;   // uniform across the block

        // Stage this 32x128 X chunk into LDS as bf16 (zero-pad past nrows).
        for (int i = tid; i < 32 * KDIM; i += 256) {
            int r = i >> 7, k = i & 127;
            long row = row0blk + r;
            float v = (row < nrows) ? X[row * KDIM + k] : 0.0f;
            sA[r * LDA + k] = (__bf16)v;
        }
        __syncthreads();

        v8f c0 = {}; v8f c1 = {};

        #pragma unroll
        for (int kk = 0; kk < 4; ++kk) {
            const int k0 = kk * 32;
            V16U a, b0, b1;
            // A 16x32: lanes<16 hold K k0..k0+15, lanes>=16 +16
            const __bf16* ap = &sA[(rt * 16 + l16) * LDA + k0 + 16 * half];
            a.h.lo = *(const v8bf*)ap;
            a.h.hi = *(const v8bf*)(ap + 8);
            // B 32x16: lane = column; lanes<16 hold K k0..k0+15, lanes>=16 +16
            const __bf16* bp0 = &sW[(n0a + l16) * LDA + k0 + 16 * half];
            b0.h.lo = *(const v8bf*)bp0;
            b0.h.hi = *(const v8bf*)(bp0 + 8);
            const __bf16* bp1 = &sW[(n0b + l16) * LDA + k0 + 16 * half];
            b1.h.lo = *(const v8bf*)bp1;
            b1.h.hi = *(const v8bf*)(bp1 + 8);

            c0 = __builtin_amdgcn_wmma_f32_16x16x32_bf16(false, a.v, false, b0.v,
                                                         (short)0, c0, false, false);
            c1 = __builtin_amdgcn_wmma_f32_16x16x32_bf16(false, a.v, false, b1.v,
                                                         (short)0, c1, false, false);
        }

        // Prefetch the next chunk's X rows while the WMMAs drain
        // (emits global_prefetch_b8; 32 rows x 512B = 256 cachelines / 256 thr).
        if (chunk < 3) {
            long prow = row0blk + 32 + (tid >> 3);
            if (prow < nrows)
                __builtin_prefetch(&X[prow * KDIM + (size_t)(tid & 7) * 16], 0, 1);
        }

        // C/D layout: VGPR j -> row (j + 8*half), col = n0 + (lane&15)
        const long row0 = row0blk + rt * 16;
        #pragma unroll
        for (int j = 0; j < 8; ++j) {
            long row = row0 + j + 8 * half;
            if (row >= nrows) continue;
            {
                float v = c0[j] + biasA;
                if (BETA) v += Y[row * KDIM + n0a + l16];
                Y[row * KDIM + n0a + l16] = v;
            }
            {
                float v = c1[j] + biasB;
                if (BETA) v += Y[row * KDIM + n0b + l16];
                Y[row * KDIM + n0b + l16] = v;
            }
        }
        __syncthreads();   // sA reused next chunk
    }
}

// ---------------------------------------------------------------------------
// Fused edge kernel: one wave per edge.
//   e_  = ea[e] @ We + be            (ED=16 contraction, We cached in LDS)
//   g   = sigmoid(k[dst] + q[src] + e_)
//   agg[dst] += g * v[src]          (fp32 global atomics = segment_sum)
// ---------------------------------------------------------------------------
__global__ __launch_bounds__(256)
void edge_gate_scatter(const int* __restrict__ ei, const float* __restrict__ ea,
                       const float* __restrict__ We, const float* __restrict__ be,
                       const float* __restrict__ kbuf, const float* __restrict__ qbuf,
                       const float* __restrict__ vbuf, float* __restrict__ agg,
                       int E)
{
    __shared__ float sWe[EDIM * KDIM];   // 8 KB
    __shared__ float sbe[KDIM];

    const int tid = threadIdx.x;
    for (int i = tid; i < EDIM * KDIM; i += 256) sWe[i] = We[i];
    if (tid < KDIM) sbe[tid] = be[tid];
    __syncthreads();

    const int wave = tid >> 5;
    const int lane = tid & 31;
    const long e   = (long)blockIdx.x * 8 + wave;
    if (e >= E) return;

    const int si = ei[e];
    const int di = ei[(size_t)E + e];

    // Broadcast ea[e][0..15] across the wave.
    float eav = (lane < EDIM) ? ea[(size_t)e * EDIM + lane] : 0.0f;
    float eam[EDIM];
    #pragma unroll
    for (int m = 0; m < EDIM; ++m) eam[m] = __shfl(eav, m, 32);

    const size_t kb = (size_t)di * KDIM;
    const size_t qb = (size_t)si * KDIM;

    #pragma unroll
    for (int j = 0; j < 4; ++j) {
        const int c = lane + 32 * j;
        float ev = sbe[c];
        #pragma unroll
        for (int m = 0; m < EDIM; ++m) ev += eam[m] * sWe[m * KDIM + c];
        float g = kbuf[kb + c] + qbuf[qb + c] + ev;
        g = 1.0f / (1.0f + __expf(-g));
        atomicAdd(&agg[kb + c], g * vbuf[qb + c]);
    }
}

// ---------------------------------------------------------------------------
// Per-batch mean/max pooling over concat(gPQ, gPV, gSL) -> gnode[B,2,128]
// ---------------------------------------------------------------------------
__global__ __launch_bounds__(128)
void gnode_reduce(const float* __restrict__ gPQ, const float* __restrict__ gPV,
                  const float* __restrict__ gSL, float* __restrict__ gnode)
{
    const int b = blockIdx.x, c = threadIdx.x;
    float s = 0.0f, mx = -3.402823466e38f;
    for (int i = 0; i < 1024; ++i) {
        float v = gPQ[((size_t)b * 1024 + i) * KDIM + c];
        s += v; mx = fmaxf(mx, v);
    }
    for (int i = 0; i < 256; ++i) {
        float v = gPV[((size_t)b * 256 + i) * KDIM + c];
        s += v; mx = fmaxf(mx, v);
    }
    {
        float v = gSL[(size_t)b * KDIM + c];
        s += v; mx = fmaxf(mx, v);
    }
    gnode[((size_t)b * 2 + 0) * KDIM + c] = s * (1.0f / 1281.0f);
    gnode[((size_t)b * 2 + 1) * KDIM + c] = mx;
}

// ---------------------------------------------------------------------------
// Cross-attention core: 8 heads x DK=16, exactly 2 keys (mean/max tokens).
// One 128-thread block per query row; head dot-products via shfl_xor(w=16).
// ---------------------------------------------------------------------------
__global__ __launch_bounds__(128)
void attn_core(const float* __restrict__ q, const float* __restrict__ k2,
               const float* __restrict__ v2, float* __restrict__ out, int N1)
{
    const long row = blockIdx.x;
    const int  c   = threadIdx.x;
    const int  b   = (int)(row / N1);

    const float qv = q[(size_t)row * KDIM + c];
    const size_t g0 = ((size_t)b * 2 + 0) * KDIM + c;
    const size_t g1 = ((size_t)b * 2 + 1) * KDIM + c;
    float p0 = qv * k2[g0];
    float p1 = qv * k2[g1];
    #pragma unroll
    for (int off = 8; off; off >>= 1) {
        p0 += __shfl_xor(p0, off, 16);
        p1 += __shfl_xor(p1, off, 16);
    }
    const float l0 = p0 * 0.25f, l1 = p1 * 0.25f;   // 1/sqrt(DK)=0.25
    const float m  = fmaxf(l0, l1);
    const float e0 = __expf(l0 - m), e1 = __expf(l1 - m);
    const float inv = 1.0f / (e0 + e1);
    out[(size_t)row * KDIM + c] = (e0 * inv) * v2[g0] + (e1 * inv) * v2[g1];
}

// ---------------------------------------------------------------------------
// LayerNorm( x + ca ) over 128 channels -> out
// ---------------------------------------------------------------------------
__global__ __launch_bounds__(128)
void resid_layernorm(const float* __restrict__ x, const float* __restrict__ ca,
                     const float* __restrict__ g, const float* __restrict__ b,
                     float* __restrict__ out)
{
    __shared__ float red[4];
    const long row = blockIdx.x;
    const int  c   = threadIdx.x;
    const float h  = x[(size_t)row * KDIM + c] + ca[(size_t)row * KDIM + c];

    float s = h;
    #pragma unroll
    for (int off = 16; off; off >>= 1) s += __shfl_xor(s, off, 32);
    if ((c & 31) == 0) red[c >> 5] = s;
    __syncthreads();
    const float mu = (red[0] + red[1] + red[2] + red[3]) * (1.0f / 128.0f);
    __syncthreads();

    const float d = h - mu;
    float v = d * d;
    #pragma unroll
    for (int off = 16; off; off >>= 1) v += __shfl_xor(v, off, 32);
    if ((c & 31) == 0) red[c >> 5] = v;
    __syncthreads();
    const float var = (red[0] + red[1] + red[2] + red[3]) * (1.0f / 128.0f);

    out[(size_t)row * KDIM + c] = d * rsqrtf(var + 1e-5f) * g[c] + b[c];
}

// ---------------------------------------------------------------------------
// ReLU (optionally writing a second destination, used for the Slack output)
// ---------------------------------------------------------------------------
__global__ void relu_copy(const float* __restrict__ in, float* __restrict__ out,
                          float* __restrict__ out2, size_t n)
{
    size_t i = (size_t)blockIdx.x * blockDim.x + threadIdx.x;
    if (i < n) {
        float v = fmaxf(in[i], 0.0f);
        out[i] = v;
        if (out2) out2[i] = v;
    }
}

// ---------------------------------------------------------------------------
// Host-side orchestration
// ---------------------------------------------------------------------------
extern "C" void kernel_launch(void* const* d_in, const int* in_sizes, int n_in,
                              void* d_out, int out_size, void* d_ws, size_t ws_size,
                              hipStream_t stream)
{
    (void)in_sizes; (void)n_in; (void)out_size; (void)ws_size;

    static const int ESRC[8] = {0, 0, 0, 1, 1, 1, 2, 2};
    static const int EDST[8] = {0, 1, 2, 0, 1, 2, 0, 1};
    static const int ECNT[8] = {262144, 131072, 4096, 131072, 65536, 2048, 4096, 2048};
    static const int NNODE[3] = {NPQ_TOT, NPV_TOT, NSL_TOT};

    const float* xin[3] = {(const float*)d_in[0], (const float*)d_in[1], (const float*)d_in[2]};
    const int*   ei[8];
    const float* ea[8];
    for (int t = 0; t < 8; ++t) {
        ei[t] = (const int*)  d_in[3 + 2 * t];
        ea[t] = (const float*)d_in[4 + 2 * t];
    }
    const float* Wk    = (const float*)d_in[19];
    const float* bk    = (const float*)d_in[20];
    const float* Wq    = (const float*)d_in[21];
    const float* bq    = (const float*)d_in[22];
    const float* Wv    = (const float*)d_in[23];
    const float* bv    = (const float*)d_in[24];
    const float* Ws    = (const float*)d_in[25];
    const float* bconv = (const float*)d_in[26];
    const float* We    = (const float*)d_in[27];
    const float* be    = (const float*)d_in[28];
    const float* Wgt   = (const float*)d_in[29];
    const float* bgt   = (const float*)d_in[30];
    const float* Wq1   = (const float*)d_in[31];
    const float* Wk2   = (const float*)d_in[32];
    const float* Wv2   = (const float*)d_in[33];
    const float* Wo    = (const float*)d_in[34];
    const float* bo    = (const float*)d_in[35];
    const float* ln_g  = (const float*)d_in[36];
    const float* ln_b  = (const float*)d_in[37];

    // Workspace carve-up (floats)
    float* w = (float*)d_ws;
    size_t o = 0;
    float* cur[3]; float* nw[3];
    for (int d = 0; d < 3; ++d) { cur[d] = w + o; o += (size_t)NNODE[d] * KDIM; }
    for (int d = 0; d < 3; ++d) { nw[d]  = w + o; o += (size_t)NNODE[d] * KDIM; }
    float* bufK  = w + o; o += (size_t)NPQ_TOT * KDIM;
    float* bufQ  = w + o; o += (size_t)NPQ_TOT * KDIM;
    float* bufV  = w + o; o += (size_t)NPQ_TOT * KDIM;
    float* gnode = w + o; o += (size_t)64 * 2 * KDIM;
    float* k2b   = w + o; o += (size_t)64 * 2 * KDIM;
    float* v2b   = w + o; o += (size_t)64 * 2 * KDIM;

    float* outPQ = (float*)d_out;
    float* outPV = outPQ + (size_t)NPQ_TOT * KDIM;
    float* outSL = outPV + (size_t)NPV_TOT * KDIM;

    auto gemm = [&](const float* X, const float* Wm, const float* bias, float* Y,
                    int nrows, int beta) {
        dim3 g((nrows + 127) / 128);
        if (beta)
            gemm128_wmma<1, 1><<<g, 256, 0, stream>>>(X, Wm, bias, Y, nrows);
        else if (bias)
            gemm128_wmma<0, 1><<<g, 256, 0, stream>>>(X, Wm, bias, Y, nrows);
        else
            gemm128_wmma<0, 0><<<g, 256, 0, stream>>>(X, Wm, nullptr, Y, nrows);
    };

    // --- Global tokens (from the ORIGINAL inputs) ---
    gemm(xin[0], Wgt, bgt, bufK, NNODE[0], 0);
    gemm(xin[1], Wgt, bgt, bufQ, NNODE[1], 0);
    gemm(xin[2], Wgt, bgt, bufV, NNODE[2], 0);
    gnode_reduce<<<64, 128, 0, stream>>>(bufK, bufQ, bufV, gnode);

    // --- Two HeteroConv layers ---
    for (int l = 0; l < 2; ++l) {
        const float* xc[3];
        for (int d = 0; d < 3; ++d) xc[d] = (l == 0) ? xin[d] : cur[d];

        // new[d] = sum_t ( x[d] @ Ws[l,t] + bconv[l,t] )  (t<3 is first per dest)
        for (int t = 0; t < 8; ++t) {
            const size_t wo = (size_t)(l * 8 + t) * KDIM * KDIM;
            const size_t bb = (size_t)(l * 8 + t) * KDIM;
            gemm(xc[EDST[t]], Ws + wo, bconv + bb, nw[EDST[t]], NNODE[EDST[t]],
                 (t >= 3) ? 1 : 0);
        }
        // Per edge type: k/q/v GEMMs then fused gate+scatter
        for (int t = 0; t < 8; ++t) {
            const int s = ESRC[t], d = EDST[t], E = ECNT[t];
            const size_t wo = (size_t)(l * 8 + t) * KDIM * KDIM;
            const size_t bb = (size_t)(l * 8 + t) * KDIM;
            const size_t we = (size_t)(l * 8 + t) * EDIM * KDIM;
            gemm(xc[d], Wk + wo, bk + bb, bufK, NNODE[d], 0);
            gemm(xc[s], Wq + wo, bq + bb, bufQ, NNODE[s], 0);
            gemm(xc[s], Wv + wo, bv + bb, bufV, NNODE[s], 0);
            edge_gate_scatter<<<dim3((E + 7) / 8), 256, 0, stream>>>(
                ei[t], ea[t], We + we, be + bb, bufK, bufQ, bufV, nw[d], E);
        }
        // ReLU -> cur (Slack final output written directly after layer 2)
        for (int d = 0; d < 3; ++d) {
            size_t n = (size_t)NNODE[d] * KDIM;
            float* second = (l == 1 && d == 2) ? outSL : nullptr;
            relu_copy<<<dim3((unsigned)((n + 255) / 256)), 256, 0, stream>>>(
                nw[d], cur[d], second, n);
        }
    }

    // --- Cross-attention vs global tokens + residual + LayerNorm ---
    gemm(gnode, Wk2, nullptr, k2b, 128, 0);
    gemm(gnode, Wv2, nullptr, v2b, 128, 0);

    // PQ
    gemm(cur[0], Wq1, nullptr, bufK, NNODE[0], 0);
    attn_core<<<dim3(NPQ_TOT), 128, 0, stream>>>(bufK, k2b, v2b, bufQ, 1024);
    gemm(bufQ, Wo, bo, bufV, NNODE[0], 0);
    resid_layernorm<<<dim3(NPQ_TOT), 128, 0, stream>>>(cur[0], bufV, ln_g, ln_b, outPQ);

    // PV
    gemm(cur[1], Wq1, nullptr, bufK, NNODE[1], 0);
    attn_core<<<dim3(NPV_TOT), 128, 0, stream>>>(bufK, k2b, v2b, bufQ, 256);
    gemm(bufQ, Wo, bo, bufV, NNODE[1], 0);
    resid_layernorm<<<dim3(NPV_TOT), 128, 0, stream>>>(cur[1], bufV, ln_g, ln_b, outPV);
}
